// MultiStreamCNN_42219528520020
// MI455X (gfx1250) — matrix operand order repaired and checked by reading, not compiled, not run on hardware
//
#include <hip/hip_runtime.h>
#include <hip/hip_bf16.h>

typedef unsigned short u16;
typedef __attribute__((ext_vector_type(16))) __bf16 v16bf;
typedef __attribute__((ext_vector_type(8)))  float  v8f;

union Frag {
    v16bf v;
    uint4 q[2];
};

__device__ __forceinline__ u16 f2bf(float f) {
    union { float f; unsigned int u; } c; c.f = f;
    unsigned int u = c.u;
    unsigned int r = (u + 0x7FFFu + ((u >> 16) & 1u)) >> 16;
    return (u16)r;
}
__device__ __forceinline__ float bf2f(u16 h) {
    union { unsigned int u; float f; } c; c.u = ((unsigned int)h) << 16;
    return c.f;
}

// ---------------------------------------------------------------------------
// Kernel 0: pack weights fp32 OIHW -> bf16 GEMM-friendly layouts; zero mean buf.
//   conv1: [oc][k=ic*9+tap, padded to 32]      (32x32 per stream)
//   conv2: [tap][oc][ic]                        (9x64x32 per stream)
//   fuse : [tap][oc][ic]                        (9x128x128)
// ---------------------------------------------------------------------------
__global__ __launch_bounds__(256) void pack_weights(
    const float* __restrict__ w1r, const float* __restrict__ w1i,
    const float* __restrict__ w2r, const float* __restrict__ w2i,
    const float* __restrict__ wf,
    u16* __restrict__ p1r, u16* __restrict__ p1i,
    u16* __restrict__ pr, u16* __restrict__ pi, u16* __restrict__ pf,
    float* __restrict__ meanbuf)
{
    int t = blockIdx.x * 256 + threadIdx.x;
    if (t < 1024) meanbuf[t] = 0.f;
    if (t < 1024) {   // conv1 pack: 32 oc x 32 k (27 real)
        int kk = t & 31, oc = t >> 5;
        u16 vr = 0, vi = 0;
        if (kk < 27) {
            int ic = kk / 9, tap = kk % 9;
            vr = f2bf(w1r[(oc * 3 + ic) * 9 + tap]);
            vi = f2bf(w1i[(oc * 3 + ic) * 9 + tap]);
        }
        p1r[t] = vr; p1i[t] = vi;
    }
    if (t < 18432) {  // 9 * 64 * 32
        int ic = t & 31, oc = (t >> 5) & 63, tap = t >> 11;
        pr[t] = f2bf(w2r[(oc * 32 + ic) * 9 + tap]);
        pi[t] = f2bf(w2i[(oc * 32 + ic) * 9 + tap]);
    }
    if (t < 147456) { // 9 * 128 * 128
        int ic = t & 127, oc = (t >> 7) & 127, tap = t >> 14;
        pf[t] = f2bf(wf[(oc * 128 + ic) * 9 + tap]);
    }
}

// ---------------------------------------------------------------------------
// Kernel 1: conv 3x3 (3->32) via WMMA (im2col in LDS, K=27 padded to 32)
//           + bias + ReLU + fused 2x2 maxpool. fp32 NCHW in, bf16 NHWC out.
// Block = 128 threads (4 waves) covers one pooled row segment of 16 px:
//   pre-pool region 2 rows x 32 px. Each wave: 16 px x 32 oc = 2 WMMAs.
// ---------------------------------------------------------------------------
__global__ __launch_bounds__(128) void conv1_wmma_pool(
    const float* __restrict__ x, const u16* __restrict__ pw,
    const float* __restrict__ bias, u16* __restrict__ out)
{
    __shared__ __align__(16) u16 im2col[64 * 32]; // 4 KB: 64 px x 32 K
    __shared__ __align__(16) u16 convo[64 * 32];  // 4 KB: pre-pool results

    const int blk = blockIdx.x;        // b(8) * py(256) * xt(16)
    const int xt  = blk & 15;
    const int py  = (blk >> 4) & 255;
    const int b   = blk >> 12;
    const int y0  = 2 * py;
    const int x0g = 32 * xt;

    // ---- stage im2col: thread owns one K column (k = tid&31), 16 pixels ----
    const int k = threadIdx.x & 31;
    int ic = 0, ky = 0, kx = 0;
    const bool kvalid = (k < 27);
    if (kvalid) { ic = k / 9; int tap = k % 9; ky = tap / 3; kx = tap % 3; }
    const float* xsrc = x + (size_t)(b * 3 + ic) * 512 * 512;
#pragma unroll
    for (int e = 0; e < 16; ++e) {
        int p   = (threadIdx.x >> 5) + e * 4;   // 0..63
        int row = p >> 5, xx = p & 31;
        int gy = y0 + row + ky - 1;
        int gx = x0g + xx + kx - 1;
        float v = 0.f;
        if (kvalid && gy >= 0 && gy < 512 && gx >= 0 && gx < 512)
            v = xsrc[(size_t)gy * 512 + gx];
        im2col[p * 32 + k] = f2bf(v);
    }
    __syncthreads();

    // ---- WMMA: wave w handles pixels [16w, 16w+16) ----
    const int lane  = threadIdx.x & 31;
    const int wave  = threadIdx.x >> 5;
    const int m     = lane & 15;
    const int kh    = lane >> 4;
    const int pbase = wave * 16;

    Frag A;
    const u16* rp = &im2col[(pbase + m) * 32];
    A.q[0] = *(const uint4*)(rp + kh * 8);
    A.q[1] = *(const uint4*)(rp + 16 + kh * 8);

    v8f acc[2];
#pragma unroll
    for (int nt = 0; nt < 2; ++nt)
#pragma unroll
        for (int e = 0; e < 8; ++e) acc[nt][e] = 0.f;

#pragma unroll
    for (int nt = 0; nt < 2; ++nt) {
        Frag B;
        const u16* wp = pw + ((nt * 16 + m) << 5);
        B.q[0] = *(const uint4*)(wp + kh * 8);
        B.q[1] = *(const uint4*)(wp + 16 + kh * 8);
        acc[nt] = __builtin_amdgcn_wmma_f32_16x16x32_bf16(
            false, A.v, false, B.v, (short)0, acc[nt], false, false);
    }

    // bias + ReLU into LDS pre-pool tile
#pragma unroll
    for (int nt = 0; nt < 2; ++nt) {
        const int oc = nt * 16 + m;
        const float bb = bias[oc];
#pragma unroll
        for (int r = 0; r < 8; ++r) {
            const int mm = (lane < 16) ? r : (r + 8);
            convo[(pbase + mm) * 32 + oc] = f2bf(fmaxf(acc[nt][r] + bb, 0.f));
        }
    }
    __syncthreads();

    // ---- 2x2 maxpool: 16 pooled px x 32 oc, NHWC bf16 out ----
#pragma unroll
    for (int e = 0; e < 4; ++e) {
        int idx = threadIdx.x + e * 128;   // 0..511
        int oc  = idx & 31;
        int pxl = idx >> 5;                // pooled x local 0..15
        int xA  = 2 * pxl;
        float v00 = bf2f(convo[xA * 32 + oc]);
        float v01 = bf2f(convo[(xA + 1) * 32 + oc]);
        float v10 = bf2f(convo[(32 + xA) * 32 + oc]);
        float v11 = bf2f(convo[(32 + xA + 1) * 32 + oc]);
        float mx  = fmaxf(fmaxf(v00, v01), fmaxf(v10, v11));
        out[((size_t)(b * 256 + py) * 256 + (16 * xt + pxl)) * 32 + oc] = f2bf(mx);
    }
}

// ---------------------------------------------------------------------------
// Kernel 2: conv 3x3 (32->64) implicit GEMM via bf16 WMMA + bias + ReLU.
// Weights staged in LDS once per block; A from global NHWC bf16.
// ---------------------------------------------------------------------------
__global__ __launch_bounds__(256) void conv2_wmma(
    const u16* __restrict__ in, const u16* __restrict__ pw,
    const float* __restrict__ bias, u16* __restrict__ out)
{
    __shared__ __align__(16) u16 sw[18432];   // 36 KB packed weights
    for (int i = threadIdx.x; i < 2304; i += 256)
        ((uint4*)sw)[i] = ((const uint4*)pw)[i];
    __syncthreads();

    const int lane = threadIdx.x & 31;
    const int wave = threadIdx.x >> 5;
    const int tile = blockIdx.x * 8 + wave;      // 32768 tiles
    const int b    = tile >> 12;
    const int rem  = tile & 4095;
    const int y    = rem >> 4;
    const int x0   = (rem & 15) << 4;
    const int m    = lane & 15;
    const int kh   = lane >> 4;
    const uint4 z4 = make_uint4(0u, 0u, 0u, 0u);

    v8f acc[4];
#pragma unroll
    for (int nt = 0; nt < 4; ++nt)
#pragma unroll
        for (int e = 0; e < 8; ++e) acc[nt][e] = 0.f;

    const int xm = x0 + m;
    __builtin_prefetch(in + (((size_t)(b * 256 + y) * 256 + xm) << 5), 0, 3);

#pragma unroll
    for (int ky = 0; ky < 3; ++ky) {
        const int iy   = y + ky - 1;
        const bool yok = (iy >= 0) && (iy < 256);
#pragma unroll
        for (int kx = 0; kx < 3; ++kx) {
            const int ix  = xm + kx - 1;
            const bool ok = yok && (ix >= 0) && (ix < 256);
            Frag A;
            if (ok) {
                const u16* rp = in + (((size_t)(b * 256 + iy) * 256 + ix) << 5);
                A.q[0] = *(const uint4*)(rp + kh * 8);
                A.q[1] = *(const uint4*)(rp + 16 + kh * 8);
            } else {
                A.q[0] = z4; A.q[1] = z4;
            }
            const int tap = ky * 3 + kx;
#pragma unroll
            for (int nt = 0; nt < 4; ++nt) {
                Frag B;
                const u16* wp = &sw[(tap * 64 + nt * 16 + m) << 5];
                B.q[0] = *(const uint4*)(wp + kh * 8);
                B.q[1] = *(const uint4*)(wp + 16 + kh * 8);
                acc[nt] = __builtin_amdgcn_wmma_f32_16x16x32_bf16(
                    false, A.v, false, B.v, (short)0, acc[nt], false, false);
            }
        }
    }
#pragma unroll
    for (int nt = 0; nt < 4; ++nt) {
        const int oc = nt * 16 + m;
        const float bb = bias[oc];
#pragma unroll
        for (int r = 0; r < 8; ++r) {
            const int mm = (lane < 16) ? r : (r + 8);
            const float v = fmaxf(acc[nt][r] + bb, 0.f);
            out[(((size_t)(b * 256 + y) * 256 + (x0 + mm)) << 6) + oc] = f2bf(v);
        }
    }
}

// ---------------------------------------------------------------------------
// Kernel 3: 2x2 maxpool NHWC (8,256,256,64) -> concat buffer (8,128,128,128)
// ---------------------------------------------------------------------------
__global__ __launch_bounds__(256) void pool2_concat(
    const u16* __restrict__ in, u16* __restrict__ cat, int cbase)
{
    int t  = blockIdx.x * 256 + threadIdx.x;   // 8*128*128*64
    int ch = t & 63;
    int px = (t >> 6) & 127;
    int py = (t >> 13) & 127;
    int b  = t >> 20;
    size_t base = ((size_t)(b * 256 + 2 * py) * 256 + 2 * px) * 64 + ch;
    float v00 = bf2f(in[base]);
    float v01 = bf2f(in[base + 64]);
    float v10 = bf2f(in[base + 16384]);
    float v11 = bf2f(in[base + 16384 + 64]);
    float mx  = fmaxf(fmaxf(v00, v01), fmaxf(v10, v11));
    cat[((size_t)(b * 128 + py) * 128 + px) * 128 + cbase + ch] = f2bf(mx);
}

// ---------------------------------------------------------------------------
// Kernel 4: fuse conv 3x3 (128->128) WMMA + bias + ReLU + partial global mean.
// ---------------------------------------------------------------------------
__global__ __launch_bounds__(256) void fuse_wmma_mean(
    const u16* __restrict__ in, const u16* __restrict__ pw,
    const float* __restrict__ bias, float* __restrict__ meanbuf)
{
    __shared__ float shacc[128];
    if (threadIdx.x < 128) shacc[threadIdx.x] = 0.f;
    __syncthreads();

    const int lane   = threadIdx.x & 31;
    const int wave   = threadIdx.x >> 5;
    const int tile   = blockIdx.x * 4 + (wave >> 1);   // 8192 tiles
    const int ocBase = (wave & 1) * 64;
    const int b      = tile >> 10;
    const int rem    = tile & 1023;
    const int y      = rem >> 3;
    const int x0     = (rem & 7) << 4;
    const int m      = lane & 15;
    const int kh     = lane >> 4;
    const uint4 z4   = make_uint4(0u, 0u, 0u, 0u);

    v8f acc[4];
#pragma unroll
    for (int nt = 0; nt < 4; ++nt)
#pragma unroll
        for (int e = 0; e < 8; ++e) acc[nt][e] = 0.f;

    __builtin_prefetch(in + ((size_t)(b * 128 + y) * 128 + x0 + m) * 128, 0, 3);

#pragma unroll
    for (int ky = 0; ky < 3; ++ky) {
        const int iy   = y + ky - 1;
        const bool yok = (iy >= 0) && (iy < 128);
#pragma unroll
        for (int kx = 0; kx < 3; ++kx) {
            const int ix  = x0 + m + kx - 1;
            const bool ok = yok && (ix >= 0) && (ix < 128);
            const int tap = ky * 3 + kx;
            const u16* rp = in + ((size_t)(b * 128 + iy) * 128 + ix) * 128;
            for (int c = 0; c < 4; ++c) {
                Frag A;
                if (ok) {
                    A.q[0] = *(const uint4*)(rp + c * 32 + kh * 8);
                    A.q[1] = *(const uint4*)(rp + c * 32 + 16 + kh * 8);
                } else {
                    A.q[0] = z4; A.q[1] = z4;
                }
#pragma unroll
                for (int nt = 0; nt < 4; ++nt) {
                    Frag B;
                    const u16* wp = pw + ((size_t)(tap * 128 + ocBase + nt * 16 + m) << 7) + c * 32;
                    B.q[0] = *(const uint4*)(wp + kh * 8);
                    B.q[1] = *(const uint4*)(wp + 16 + kh * 8);
                    acc[nt] = __builtin_amdgcn_wmma_f32_16x16x32_bf16(
                        false, A.v, false, B.v, (short)0, acc[nt], false, false);
                }
            }
        }
    }
#pragma unroll
    for (int nt = 0; nt < 4; ++nt) {
        const int oc = ocBase + nt * 16 + m;
        const float bb = bias[oc];
        float s = 0.f;
#pragma unroll
        for (int r = 0; r < 8; ++r) s += fmaxf(acc[nt][r] + bb, 0.f);
        atomicAdd(&shacc[oc], s);
    }
    __syncthreads();
    if (threadIdx.x < 128) {
        const int bb = (blockIdx.x * 4) >> 10;  // whole block is one batch
        atomicAdd(&meanbuf[bb * 128 + threadIdx.x], shacc[threadIdx.x]);
    }
}

// ---------------------------------------------------------------------------
// Kernel 5: det head (128->18), sigmoid, NMS over 3 boxes, write outputs.
// d_out layout: p[144] | boxes[96] | scores[24] | keep[24]
// ---------------------------------------------------------------------------
__device__ __forceinline__ float iou_f(const float* A, const float* B)
{
    float aA = (A[2] - A[0]) * (A[3] - A[1]);
    float aB = (B[2] - B[0]) * (B[3] - B[1]);
    float ix1 = fmaxf(A[0], B[0]), iy1 = fmaxf(A[1], B[1]);
    float ix2 = fminf(A[2], B[2]), iy2 = fminf(A[3], B[3]);
    float inter = fmaxf(ix2 - ix1, 0.f) * fmaxf(iy2 - iy1, 0.f);
    return inter / (aA + aB - inter);
}

__global__ __launch_bounds__(256) void finalize_det(
    const float* __restrict__ meanbuf, const float* __restrict__ dw,
    const float* __restrict__ db, float* __restrict__ out)
{
    __shared__ float lp[144];
    const int t = threadIdx.x;
    if (t < 144) {
        const int b = t / 18, o = t % 18;
        const float* f = meanbuf + b * 128;
        float s = db[o];
        for (int c = 0; c < 128; ++c)
            s = fmaf(dw[o * 128 + c], f[c] * (1.f / 16384.f), s);
        lp[t] = s;
        out[t] = s;
    }
    __syncthreads();
    if (t < 8) {
        const int b = t;
        float bx[3][4], sc[3], key[3];
        bool ck[3];
        for (int a = 0; a < 3; ++a) {
            for (int k = 0; k < 4; ++k) bx[a][k] = lp[b * 18 + a * 6 + k];
            float lg = lp[b * 18 + a * 6 + 4];
            sc[a] = 1.f / (1.f + __expf(-lg));
            ck[a] = sc[a] > 0.5f;
            key[a] = ck[a] ? sc[a] : -3.4e38f;
        }
        int ord[3] = {0, 1, 2};
        if (key[ord[0]] < key[ord[1]]) { int q = ord[0]; ord[0] = ord[1]; ord[1] = q; }
        if (key[ord[1]] < key[ord[2]]) { int q = ord[1]; ord[1] = ord[2]; ord[2] = q; }
        if (key[ord[0]] < key[ord[1]]) { int q = ord[0]; ord[0] = ord[1]; ord[1] = q; }
        bool sup[3]  = {false, false, false};
        bool keep[3] = {false, false, false};
        for (int i = 0; i < 3; ++i) {
            if (!sup[i]) {
                keep[ord[i]] = ck[ord[i]];
                for (int j = i + 1; j < 3; ++j)
                    if (iou_f(bx[ord[i]], bx[ord[j]]) > 0.5f) sup[j] = true;
            }
        }
        for (int a = 0; a < 3; ++a) {
            for (int k = 0; k < 4; ++k) out[144 + b * 12 + a * 4 + k] = bx[a][k];
            out[240 + b * 3 + a] = sc[a];
            out[264 + b * 3 + a] = keep[a] ? 1.f : 0.f;
        }
    }
}

// ---------------------------------------------------------------------------
extern "C" void kernel_launch(void* const* d_in, const int* in_sizes, int n_in,
                              void* d_out, int out_size, void* d_ws, size_t ws_size,
                              hipStream_t stream)
{
    const float* x   = (const float*)d_in[0];
    const float* rw1 = (const float*)d_in[1];
    const float* rb1 = (const float*)d_in[2];
    const float* rw2 = (const float*)d_in[3];
    const float* rb2 = (const float*)d_in[4];
    const float* iw1 = (const float*)d_in[5];
    const float* ib1 = (const float*)d_in[6];
    const float* iw2 = (const float*)d_in[7];
    const float* ib2 = (const float*)d_in[8];
    const float* fw  = (const float*)d_in[9];
    const float* fb  = (const float*)d_in[10];
    const float* dw  = (const float*)d_in[11];
    const float* db  = (const float*)d_in[12];
    float* out = (float*)d_out;

    char* ws = (char*)d_ws;
    size_t off = 0;
    auto alloc = [&](size_t bytes) {
        void* p = ws + off;
        off += (bytes + 255) & ~(size_t)255;
        return p;
    };
    u16*   p1r  = (u16*)alloc(8ull * 256 * 256 * 32 * 2);   // pooled stream1 rgb, NHWC bf16
    u16*   p1i  = (u16*)alloc(8ull * 256 * 256 * 32 * 2);   // pooled stream1 ir
    u16*   c2o  = (u16*)alloc(8ull * 256 * 256 * 64 * 2);   // conv2 out (reused per stream)
    u16*   cat  = (u16*)alloc(8ull * 128 * 128 * 128 * 2);  // concat NHWC bf16
    u16*   pw1r = (u16*)alloc(32ull * 32 * 2);              // packed conv1 rgb (K padded)
    u16*   pw1i = (u16*)alloc(32ull * 32 * 2);              // packed conv1 ir
    u16*   pwr  = (u16*)alloc(9ull * 64 * 32 * 2);          // packed conv2 rgb
    u16*   pwi  = (u16*)alloc(9ull * 64 * 32 * 2);          // packed conv2 ir
    u16*   pwf  = (u16*)alloc(9ull * 128 * 128 * 2);        // packed fuse
    float* mb   = (float*)alloc(8ull * 128 * 4);            // mean accumulator

    pack_weights<<<576, 256, 0, stream>>>(rw1, iw1, rw2, iw2, fw,
                                          pw1r, pw1i, pwr, pwi, pwf, mb);

    conv1_wmma_pool<<<32768, 128, 0, stream>>>(x, pw1r, rb1, p1r);
    conv1_wmma_pool<<<32768, 128, 0, stream>>>(x, pw1i, ib1, p1i);

    conv2_wmma<<<4096, 256, 0, stream>>>(p1r, pwr, rb2, c2o);
    pool2_concat<<<32768, 256, 0, stream>>>(c2o, cat, 0);

    conv2_wmma<<<4096, 256, 0, stream>>>(p1i, pwi, ib2, c2o);
    pool2_concat<<<32768, 256, 0, stream>>>(c2o, cat, 64);

    fuse_wmma_mean<<<2048, 256, 0, stream>>>(cat, pwf, fb, mb);

    finalize_det<<<1, 256, 0, stream>>>(mb, dw, db, out);
}
